// DTNetv1_42296837931602
// MI455X (gfx1250) — compile-verified, matrix operand order checked by reading, not run on hardware
//
#include <hip/hip_runtime.h>
#include <hip/hip_bf16.h>
#include <stdint.h>

typedef __attribute__((ext_vector_type(16))) __bf16   v16bf;
typedef __attribute__((ext_vector_type(8)))  float    v8f;
typedef __attribute__((ext_vector_type(4)))  unsigned u32x4;
typedef __attribute__((ext_vector_type(8)))  int      i32x8;
typedef __attribute__((ext_vector_type(4)))  int      i32x4;

#define B_ROWS 32768
#define IN_DIM 128
#define D_DIM  511
#define A_DIM  512
#define H_DIM  1024      // 1023 padded to 1024
#define K2     2048      // 2046 padded to 2048
#define L_DIM  1024
#define NACT   16

// workspace layout (bytes)
#define OFF_XB   0ull                                   // 32768*128*2   = 8 MiB
#define OFF_W1B  (8ull*1024*1024)                       // 1024*128*2    = 256 KiB
#define OFF_B1   (OFF_W1B + 256ull*1024)                // 1024*4        = 4 KiB
#define OFF_WLB  (OFF_B1 + 4096ull)                     // 1024*2048*2   = 4 MiB
#define OFF_HHP  (OFF_WLB + 4ull*1024*1024)             // 32768*2048*2  = 128 MiB

static __device__ __forceinline__ unsigned short f2bf(float f) {
    unsigned u = __builtin_bit_cast(unsigned, f);
    u += 0x7FFFu + ((u >> 16) & 1u);       // round-to-nearest-even
    return (unsigned short)(u >> 16);
}

// ---------------- prep kernels (bf16 packing) ----------------
__global__ void prep_x(const float* __restrict__ x, unsigned short* __restrict__ xb, int n) {
    for (int i = blockIdx.x * blockDim.x + threadIdx.x; i < n; i += gridDim.x * blockDim.x)
        xb[i] = f2bf(x[i]);
}

__global__ void prep_w1(const float* __restrict__ Wd, const float* __restrict__ bd,
                        const float* __restrict__ Wa, const float* __restrict__ ba,
                        unsigned short* __restrict__ W1b, float* __restrict__ b1) {
    int i = blockIdx.x * blockDim.x + threadIdx.x;      // H_DIM*IN_DIM
    if (i >= H_DIM * IN_DIM) return;
    int j = i >> 7, k = i & 127;
    float v = 0.f;
    if (j < D_DIM)              v = Wd[j * IN_DIM + k];
    else if (j < D_DIM + A_DIM) v = Wa[(j - D_DIM) * IN_DIM + k];
    W1b[i] = f2bf(v);
    if (k == 0) {
        float b = 0.f;
        if (j < D_DIM)              b = bd[j];
        else if (j < D_DIM + A_DIM) b = ba[j - D_DIM];
        b1[j] = b;
    }
}

__global__ void prep_wl(const float* __restrict__ Wl, unsigned short* __restrict__ Wlb) {
    int i = blockIdx.x * blockDim.x + threadIdx.x;      // L_DIM*K2
    if (i >= L_DIM * K2) return;
    int l = i >> 11, c = i & 2047;
    float v = 0.f;
    if (c < 1023)                     v = Wl[l * 2046 + c];        // relu(+h) block
    else if (c >= 1024 && c < 2047)   v = Wl[l * 2046 + (c - 1)];  // relu(-h) block
    Wlb[i] = f2bf(v);
}

// ---------------- GEMM1: hh = [relu(xW1^T+b) | relu(-(...))] in bf16 ----------------
__global__ void __launch_bounds__(256) gemm1_relu(
        const unsigned short* __restrict__ xb, const unsigned short* __restrict__ W1b,
        const float* __restrict__ b1, unsigned short* __restrict__ hhp) {
    const int lane = threadIdx.x & 31;
    const int wid  = threadIdx.x >> 5;
    const int Mb = blockIdx.x * 64 + (wid & 3) * 16;
    const int Nb = blockIdx.y * 32 + (wid >> 2) * 16;
    const int mn = lane & 15;          // A: row M, B: col N
    const int hi = lane >> 4;          // K sub-chunk selector
    const unsigned short* ap = xb  + (size_t)(Mb + mn) * IN_DIM;
    const unsigned short* bp = W1b + (size_t)(Nb + mn) * IN_DIM;
    v8f c = {0.f, 0.f, 0.f, 0.f, 0.f, 0.f, 0.f, 0.f};
#pragma unroll
    for (int kb = 0; kb < IN_DIM; kb += 32) {
        const int k0 = kb + hi * 8, k1 = kb + 16 + hi * 8;
        union { uint4 q[2]; v16bf v; } A, Bm;
        A.q[0]  = *(const uint4*)(ap + k0);
        A.q[1]  = *(const uint4*)(ap + k1);
        Bm.q[0] = *(const uint4*)(bp + k0);
        Bm.q[1] = *(const uint4*)(bp + k1);
        c = __builtin_amdgcn_wmma_f32_16x16x32_bf16(false, A.v, false, Bm.v,
                                                    (short)0, c, false, false);
    }
    const float bias = b1[Nb + mn];    // each lane's 8 accs share one N column
#pragma unroll
    for (int v = 0; v < 8; ++v) {
        const int m = Mb + v + hi * 8;
        const float h = c[v] + bias;
        hhp[(size_t)m * K2 + (Nb + mn)]          = f2bf(h > 0.f ? h : 0.f);
        hhp[(size_t)m * K2 + H_DIM + (Nb + mn)]  = f2bf(h < 0.f ? -h : 0.f);
    }
}

// ---------------- GEMM2 + ragged segment-max + softmax (fused) ----------------
#define MT2       32                                  // rows per block
#define SEG_OFF   (MT2 * K2 * 2)                      // 131072 B: hh tile in LDS
#define POOL_OFF  (SEG_OFF + L_DIM * 4)               // +4 KiB seg ids
#define SMEM_BYTES (POOL_OFF + MT2 * NACT * 4)        // +2 KiB pooled maxes = 137216

__global__ void __launch_bounds__(512) gemm2_pool_softmax(
        const unsigned short* __restrict__ hhp, const unsigned short* __restrict__ Wlb,
        const int* __restrict__ seg_ids, float* __restrict__ out) {
    extern __shared__ char smem[];
    unsigned short* sA   = (unsigned short*)smem;
    int*            sSeg = (int*)(smem + SEG_OFF);
    float*          sPool = (float*)(smem + POOL_OFF);

    const int tid = threadIdx.x;
    const int lane = tid & 31, wid = tid >> 5;
    const int Mb = blockIdx.x * MT2;

    for (int i = tid; i < L_DIM; i += 512) sSeg[i] = seg_ids[i];
    if (tid < MT2 * NACT) sPool[tid] = __builtin_bit_cast(float, 0xff800000u); // -inf

    // --- TDM: async 2D tile load, 2048x32 bf16 of hh into LDS (wave 0 issues) ---
    if (wid == 0) {
        const unsigned long long ga =
            (unsigned long long)(uintptr_t)hhp + (unsigned long long)Mb * (K2 * 2);
        u32x4 g0 = {0u, 0u, 0u, 0u};
        i32x8 g1 = {0, 0, 0, 0, 0, 0, 0, 0};
        i32x4 g2 = {0, 0, 0, 0}, g3 = {0, 0, 0, 0};
        i32x8 g4 = {0, 0, 0, 0, 0, 0, 0, 0};
        g0[0] = 1u;                                         // count=1 valid descriptor
        g0[1] = 0u;                                         // lds_addr = 0 (tile at smem base)
        g0[2] = (unsigned)(ga & 0xffffffffull);             // global_addr lo
        g0[3] = (unsigned)((ga >> 32) & 0x1ffffffull) | (2u << 30);   // addr hi | type=2
        g1[0] = (int)(1u << 16);                            // data_size=1 -> 2 bytes
        g1[1] = (int)((unsigned)K2 << 16);                  // tensor_dim0 lo16 @ bits[63:48]
        g1[2] = (int)((unsigned)B_ROWS << 16);              // tensor_dim1 lo16 @ bits[95:80]
        g1[3] = (int)((unsigned)K2 << 16);                  // tile_dim0 @ bits[127:112]
        g1[4] = MT2;                                        // tile_dim1 = 32 rows
        g1[5] = K2;                                         // tensor_dim0_stride lo32
        __builtin_amdgcn_tensor_load_to_lds(g0, g1, g2, g3, g4, 0);
        __builtin_amdgcn_s_wait_tensorcnt(0);
    }
    __syncthreads();

    // 16 waves = 2(M) x 8(N); each wave owns a 16x64 tile; 2 passes cover N=1024.
    // One A-frag (2x ds_load_b128) feeds 4 WMMAs -> 2.5 load-instr per WMMA.
    const int mh = wid & 1;
    const int nw = wid >> 1;
    const int mn = lane & 15, hi = lane >> 4;
    const unsigned short* aRow = sA + (size_t)(mh * 16 + mn) * K2;
    for (int p = 0; p < 2; ++p) {
        const int Nb = p * 512 + nw * 64;
        const unsigned short* bRow0 = Wlb + (size_t)(Nb +  0 + mn) * K2;
        const unsigned short* bRow1 = Wlb + (size_t)(Nb + 16 + mn) * K2;
        const unsigned short* bRow2 = Wlb + (size_t)(Nb + 32 + mn) * K2;
        const unsigned short* bRow3 = Wlb + (size_t)(Nb + 48 + mn) * K2;
        v8f c0 = {0.f,0.f,0.f,0.f,0.f,0.f,0.f,0.f};
        v8f c1 = c0, c2 = c0, c3 = c0;
#pragma unroll 2
        for (int kb = 0; kb < K2; kb += 32) {
            const int k0 = kb + hi * 8, k1 = kb + 16 + hi * 8;
            union { uint4 q[2]; v16bf v; } A, B0, B1, B2, B3;
            A.q[0]  = *(const uint4*)(aRow + k0);      // ds_load_b128 from LDS tile
            A.q[1]  = *(const uint4*)(aRow + k1);
            B0.q[0] = *(const uint4*)(bRow0 + k0);     // global (L2-resident Wlb)
            B0.q[1] = *(const uint4*)(bRow0 + k1);
            B1.q[0] = *(const uint4*)(bRow1 + k0);
            B1.q[1] = *(const uint4*)(bRow1 + k1);
            B2.q[0] = *(const uint4*)(bRow2 + k0);
            B2.q[1] = *(const uint4*)(bRow2 + k1);
            B3.q[0] = *(const uint4*)(bRow3 + k0);
            B3.q[1] = *(const uint4*)(bRow3 + k1);
            c0 = __builtin_amdgcn_wmma_f32_16x16x32_bf16(false, A.v, false, B0.v,
                                                         (short)0, c0, false, false);
            c1 = __builtin_amdgcn_wmma_f32_16x16x32_bf16(false, A.v, false, B1.v,
                                                         (short)0, c1, false, false);
            c2 = __builtin_amdgcn_wmma_f32_16x16x32_bf16(false, A.v, false, B2.v,
                                                         (short)0, c2, false, false);
            c3 = __builtin_amdgcn_wmma_f32_16x16x32_bf16(false, A.v, false, B3.v,
                                                         (short)0, c3, false, false);
        }
        // ragged per-action max: ds float-max atomics into the 32x16 pool
#pragma unroll
        for (int nt = 0; nt < 4; ++nt) {
            const v8f c = (nt == 0) ? c0 : (nt == 1) ? c1 : (nt == 2) ? c2 : c3;
            const int seg = sSeg[Nb + nt * 16 + mn];
#pragma unroll
            for (int v = 0; v < 8; ++v) {
                const int row = mh * 16 + v + hi * 8;
                const unsigned paddr = POOL_OFF + (unsigned)(row * NACT + seg) * 4u;
                float val = c[v];
                asm volatile("ds_max_num_f32 %0, %1" :: "v"(paddr), "v"(val));
            }
        }
    }
    asm volatile("s_wait_dscnt 0x0" ::: "memory");
    __syncthreads();

    // softmax over the 16 pooled actions, one thread per row
    if (tid < MT2) {
        float mx = -3.4e38f;
#pragma unroll
        for (int a = 0; a < NACT; ++a) mx = fmaxf(mx, sPool[tid * NACT + a]);
        float e[NACT], s = 0.f;
#pragma unroll
        for (int a = 0; a < NACT; ++a) { e[a] = __expf(sPool[tid * NACT + a] - mx); s += e[a]; }
        const float inv = 1.f / s;
        float* op = out + (size_t)(Mb + tid) * NACT;
#pragma unroll
        for (int a = 0; a < NACT; ++a) op[a] = e[a] * inv;
    }
}

// ---------------- host launcher ----------------
extern "C" void kernel_launch(void* const* d_in, const int* in_sizes, int n_in,
                              void* d_out, int out_size, void* d_ws, size_t ws_size,
                              hipStream_t stream) {
    const float* x  = (const float*)d_in[0];
    const float* Wd = (const float*)d_in[1];
    const float* bd = (const float*)d_in[2];
    const float* Wa = (const float*)d_in[3];
    const float* ba = (const float*)d_in[4];
    const float* Wl = (const float*)d_in[5];
    const int*  seg = (const int*)  d_in[6];
    char* ws = (char*)d_ws;
    unsigned short* xb  = (unsigned short*)(ws + OFF_XB);
    unsigned short* W1b = (unsigned short*)(ws + OFF_W1B);
    float*          b1  = (float*)(ws + OFF_B1);
    unsigned short* Wlb = (unsigned short*)(ws + OFF_WLB);
    unsigned short* hhp = (unsigned short*)(ws + OFF_HHP);
    float* out = (float*)d_out;

    prep_x<<<4096, 256, 0, stream>>>(x, xb, B_ROWS * IN_DIM);
    prep_w1<<<(H_DIM * IN_DIM + 255) / 256, 256, 0, stream>>>(Wd, bd, Wa, ba, W1b, b1);
    prep_wl<<<(L_DIM * K2 + 255) / 256, 256, 0, stream>>>(Wl, Wlb);
    gemm1_relu<<<dim3(B_ROWS / 64, H_DIM / 32), 256, 0, stream>>>(xb, W1b, b1, hhp);
    gemm2_pool_softmax<<<B_ROWS / MT2, 512, SMEM_BYTES, stream>>>(hhp, Wlb, seg, out);
}
